// DifferentiableRecurrentNet_15736760173165
// MI455X (gfx1250) — matrix operand order repaired and checked by reading, not compiled, not run on hardware
//
#include <hip/hip_runtime.h>

#define B_  8192
#define NI_ 512
#define NH_ 2048
#define NO_ 512

typedef __attribute__((ext_vector_type(16))) __bf16 v16bf;
typedef __attribute__((ext_vector_type(8)))  __bf16 bf16x8;
typedef __attribute__((ext_vector_type(8)))  float  v8f;

union ABFrag { v16bf v; bf16x8 h[2]; };

// ---------------------------------------------------------------------------
// f32 -> bf16 conversion (grid-stride)
// ---------------------------------------------------------------------------
__global__ void cvt_f32_to_bf16(const float* __restrict__ in,
                                __bf16* __restrict__ out, size_t n) {
    size_t i      = (size_t)blockIdx.x * blockDim.x + threadIdx.x;
    size_t stride = (size_t)gridDim.x * blockDim.x;
    for (; i < n; i += stride) out[i] = (__bf16)in[i];
}

// ---------------------------------------------------------------------------
// Wave-level GEMM accumulate, 32Mx64N tile:
//   acc[mi][ns] += A[m0+mi*16 : +16, :K] * W[n0+ns*16 : +16, :K]^T
// A: row-major (M x K) bf16, W: row-major (N x K) bf16.
// Fragment layouts: 16-bit A 16x32 / B 32x16 in contiguous k = 16*half + s order; f32 C/D 16x16 per ISA.
// 8 WMMAs per 12 b128 loads per K-step of 32 (A frags reused 4x, B frags 2x).
// ---------------------------------------------------------------------------
__device__ __forceinline__ void gemm_acc(const __bf16* __restrict__ A,
                                         const __bf16* __restrict__ W,
                                         int K, int m0, int n0,
                                         int l15, int half, v8f acc[2][4]) {
    const __bf16* a0 = A + (size_t)(m0 + l15) * K + half * 16;
    const __bf16* a1 = a0 + (size_t)16 * K;
    const __bf16* b0 = W + (size_t)(n0 +  0 + l15) * K + half * 16;
    const __bf16* b1 = W + (size_t)(n0 + 16 + l15) * K + half * 16;
    const __bf16* b2 = W + (size_t)(n0 + 32 + l15) * K + half * 16;
    const __bf16* b3 = W + (size_t)(n0 + 48 + l15) * K + half * 16;

    for (int k = 0; k < K; k += 32) {
        ABFrag af0, af1;
        af0.h[0] = *(const bf16x8*)(a0 + k);       // K = k..k+7   (lanes 0-15) / k+16..k+23 (lanes 16-31)
        af0.h[1] = *(const bf16x8*)(a0 + k + 8);   // K = k+8..15  (lanes 0-15) / k+24..k+31 (lanes 16-31)
        af1.h[0] = *(const bf16x8*)(a1 + k);
        af1.h[1] = *(const bf16x8*)(a1 + k + 8);

        ABFrag f0, f1, f2, f3;
        f0.h[0] = *(const bf16x8*)(b0 + k);
        f0.h[1] = *(const bf16x8*)(b0 + k + 8);
        f1.h[0] = *(const bf16x8*)(b1 + k);
        f1.h[1] = *(const bf16x8*)(b1 + k + 8);
        f2.h[0] = *(const bf16x8*)(b2 + k);
        f2.h[1] = *(const bf16x8*)(b2 + k + 8);
        f3.h[0] = *(const bf16x8*)(b3 + k);
        f3.h[1] = *(const bf16x8*)(b3 + k + 8);

        acc[0][0] = __builtin_amdgcn_wmma_f32_16x16x32_bf16(false, af0.v, false, f0.v, (short)0, acc[0][0], false, false);
        acc[0][1] = __builtin_amdgcn_wmma_f32_16x16x32_bf16(false, af0.v, false, f1.v, (short)0, acc[0][1], false, false);
        acc[0][2] = __builtin_amdgcn_wmma_f32_16x16x32_bf16(false, af0.v, false, f2.v, (short)0, acc[0][2], false, false);
        acc[0][3] = __builtin_amdgcn_wmma_f32_16x16x32_bf16(false, af0.v, false, f3.v, (short)0, acc[0][3], false, false);
        acc[1][0] = __builtin_amdgcn_wmma_f32_16x16x32_bf16(false, af1.v, false, f0.v, (short)0, acc[1][0], false, false);
        acc[1][1] = __builtin_amdgcn_wmma_f32_16x16x32_bf16(false, af1.v, false, f1.v, (short)0, acc[1][1], false, false);
        acc[1][2] = __builtin_amdgcn_wmma_f32_16x16x32_bf16(false, af1.v, false, f2.v, (short)0, acc[1][2], false, false);
        acc[1][3] = __builtin_amdgcn_wmma_f32_16x16x32_bf16(false, af1.v, false, f3.v, (short)0, acc[1][3], false, false);
    }
}

// ---------------------------------------------------------------------------
// Hidden step: Aout = act( hr * (X@Wih^T + Aprev@Whh^T) + hb )
// Aprev == nullptr on step 1 (activations start at zero).
// Grid: (B/64, NH/256), block 256 threads = 8 waves (2 along M x 4 along N),
// block tile 64M x 256N, wave tile 32M x 64N.
// ---------------------------------------------------------------------------
__global__ __launch_bounds__(256) void hidden_step_kernel(
    const __bf16* __restrict__ Xbf,   const __bf16* __restrict__ Wih,
    const __bf16* __restrict__ Aprev, const __bf16* __restrict__ Whh,
    const float* __restrict__ resp,   const float* __restrict__ bias,
    __bf16* __restrict__ Aout) {
    const int lane = threadIdx.x & 31;
    const int wid  = threadIdx.x >> 5;
    const int l15  = lane & 15;
    const int half = lane >> 4;
    const int m0   = blockIdx.x * 64  + (wid & 1) * 32;
    const int n0   = blockIdx.y * 256 + (wid >> 1) * 64;

    v8f acc[2][4] = {};
    gemm_acc(Xbf, Wih, NI_, m0, n0, l15, half, acc);
    if (Aprev) gemm_acc(Aprev, Whh, NH_, m0, n0, l15, half, acc);

#pragma unroll
    for (int ns = 0; ns < 4; ++ns) {
        const int   n = n0 + ns * 16 + l15;
        const float r = resp[n];
        const float b = bias[n];
        const int   g = n % 3;   // 0:tanh 1:sigmoid 2:relu
#pragma unroll
        for (int mi = 0; mi < 2; ++mi) {
#pragma unroll
            for (int i = 0; i < 8; ++i) {
                const int   m   = m0 + mi * 16 + half * 8 + i;
                const float pre = r * acc[mi][ns][i] + b;
                float v;
                if (g == 0)      v = tanhf(pre);
                else if (g == 1) v = 1.0f / (1.0f + expf(-pre));
                else             v = fmaxf(pre, 0.0f);
                Aout[(size_t)m * NH_ + n] = (__bf16)v;
            }
        }
    }
}

// ---------------------------------------------------------------------------
// Output layer: out = sigmoid( or * (X@Wio^T + A@Who^T) + ob ), f32 result.
// Grid: (B/64, NO/256), block 256 threads.
// ---------------------------------------------------------------------------
__global__ __launch_bounds__(256) void output_kernel(
    const __bf16* __restrict__ Xbf, const __bf16* __restrict__ Wio,
    const __bf16* __restrict__ A,   const __bf16* __restrict__ Who,
    const float* __restrict__ resp, const float* __restrict__ bias,
    float* __restrict__ out) {
    const int lane = threadIdx.x & 31;
    const int wid  = threadIdx.x >> 5;
    const int l15  = lane & 15;
    const int half = lane >> 4;
    const int m0   = blockIdx.x * 64  + (wid & 1) * 32;
    const int n0   = blockIdx.y * 256 + (wid >> 1) * 64;

    v8f acc[2][4] = {};
    gemm_acc(Xbf, Wio, NI_, m0, n0, l15, half, acc);
    gemm_acc(A,   Who, NH_, m0, n0, l15, half, acc);

#pragma unroll
    for (int ns = 0; ns < 4; ++ns) {
        const int   n = n0 + ns * 16 + l15;
        const float r = resp[n];
        const float b = bias[n];
#pragma unroll
        for (int mi = 0; mi < 2; ++mi) {
#pragma unroll
            for (int i = 0; i < 8; ++i) {
                const int   m   = m0 + mi * 16 + half * 8 + i;
                const float pre = r * acc[mi][ns][i] + b;
                out[(size_t)m * NO_ + n] = 1.0f / (1.0f + expf(-pre));
            }
        }
    }
}

// ---------------------------------------------------------------------------
// Launch. Inputs (setup_inputs order):
// 0 inputs, 1 input_to_hidden, 2 hidden_to_hidden, 3 output_to_hidden (unused: outputs==0),
// 4 input_to_output, 5 hidden_to_output, 6 output_to_output (unused),
// 7 hidden_responses, 8 hidden_biases, 9 output_responses, 10 output_biases.
// ---------------------------------------------------------------------------
extern "C" void kernel_launch(void* const* d_in, const int* in_sizes, int n_in,
                              void* d_out, int out_size, void* d_ws, size_t ws_size,
                              hipStream_t stream) {
    (void)in_sizes; (void)n_in; (void)out_size; (void)ws_size;

    const float* X   = (const float*)d_in[0];
    const float* Wih = (const float*)d_in[1];
    const float* Whh = (const float*)d_in[2];
    const float* Wio = (const float*)d_in[4];
    const float* Who = (const float*)d_in[5];
    const float* hr  = (const float*)d_in[7];
    const float* hb  = (const float*)d_in[8];
    const float* orr = (const float*)d_in[9];
    const float* ob  = (const float*)d_in[10];
    float* out = (float*)d_out;

    char* ws = (char*)d_ws;
    size_t off = 0;
    auto alloc = [&](size_t bytes) -> void* {
        void* p = ws + off;
        off += (bytes + 255) & ~(size_t)255;
        return p;
    };
    __bf16* Xbf   = (__bf16*)alloc((size_t)B_  * NI_ * 2);
    __bf16* WihBf = (__bf16*)alloc((size_t)NH_ * NI_ * 2);
    __bf16* WhhBf = (__bf16*)alloc((size_t)NH_ * NH_ * 2);
    __bf16* WhoBf = (__bf16*)alloc((size_t)NO_ * NH_ * 2);
    __bf16* WioBf = (__bf16*)alloc((size_t)NO_ * NI_ * 2);
    __bf16* A0    = (__bf16*)alloc((size_t)B_  * NH_ * 2);
    __bf16* A1    = (__bf16*)alloc((size_t)B_  * NH_ * 2);

    // bf16 conversions
    cvt_f32_to_bf16<<<2048, 256, 0, stream>>>(X,   Xbf,   (size_t)B_  * NI_);
    cvt_f32_to_bf16<<<1024, 256, 0, stream>>>(Wih, WihBf, (size_t)NH_ * NI_);
    cvt_f32_to_bf16<<<2048, 256, 0, stream>>>(Whh, WhhBf, (size_t)NH_ * NH_);
    cvt_f32_to_bf16<<<1024, 256, 0, stream>>>(Who, WhoBf, (size_t)NO_ * NH_);
    cvt_f32_to_bf16<<<512,  256, 0, stream>>>(Wio, WioBf, (size_t)NO_ * NI_);

    dim3 blk(256);
    dim3 hgrid(B_ / 64, NH_ / 256);
    dim3 ogrid(B_ / 64, NO_ / 256);

    // 4 recurrent steps (step 1 has zero initial activations -> skip Whh phase)
    hidden_step_kernel<<<hgrid, blk, 0, stream>>>(Xbf, WihBf, nullptr, WhhBf, hr, hb, A0);
    hidden_step_kernel<<<hgrid, blk, 0, stream>>>(Xbf, WihBf, A0,      WhhBf, hr, hb, A1);
    hidden_step_kernel<<<hgrid, blk, 0, stream>>>(Xbf, WihBf, A1,      WhhBf, hr, hb, A0);
    hidden_step_kernel<<<hgrid, blk, 0, stream>>>(Xbf, WihBf, A0,      WhhBf, hr, hb, A1);

    // output layer (USE_CURRENT_ACTIVS=True -> use A1)
    output_kernel<<<ogrid, blk, 0, stream>>>(Xbf, WioBf, A1, WhoBf, orr, ob, out);
}
